// FER_GAT_41566693491250
// MI455X (gfx1250) — matrix operand order repaired and checked by reading, not compiled, not run on hardware
//
#include <hip/hip_runtime.h>
#include <hip/hip_bf16.h>

typedef __attribute__((ext_vector_type(16))) _Float16 v16h;
typedef __attribute__((ext_vector_type(8)))  float    v8f;

#define N_NODES 51
#define NPAD    64
#define HEADS   4
#define HID     64
#define OUTG    128
#define HDIM    256      // HEADS*HID
#define LSTM_B  50       // "batch" of the LSTM = F
#define SEQ     256      // time steps of the LSTM = B
#define HSZ     32
#define NIN     51
#define NST     1600     // LSTM_B*HSZ

__device__ __forceinline__ float leakyf(float x) { return x >= 0.f ? x : 0.01f * x; }
__device__ __forceinline__ float sigf(float x)   { return 1.f / (1.f + __expf(-x)); }

// ---- CDNA5 wave32 WMMA fragment loaders (16x16x32 f16, f32 accum) ----
// A: 16x32 (MxK), row-major [M][K]. Lanes 0-15: row m, K {0-7,16-23}; lanes 16-31: K {8-15,24-31}.
// Both K chunks are contiguous 8-half runs -> 2x ds_load_b128 per lane.
__device__ __forceinline__ v16h ldsA(const _Float16* src, int lda, int mof, int kof, int lane) {
  const _Float16* row = src + (mof + (lane & 15)) * lda + kof;
  int kb = (lane < 16) ? 0 : 8;
  v16h a;
#pragma unroll
  for (int v = 0; v < 8; ++v) {
    int k = (v < 4) ? (kb + 2 * v) : (16 + kb + 2 * (v - 4));
    a[2 * v]     = row[k];
    a[2 * v + 1] = row[k + 1];
  }
  return a;
}
// A fragment from global f32 row-major [M][K] (W2[o][d]) with f32->f16 convert.
__device__ __forceinline__ v16h ldgA(const float* __restrict__ src, int lda, int mof, int kof, int lane) {
  const float* row = src + (mof + (lane & 15)) * lda + kof;
  int kb = (lane < 16) ? 0 : 8;
  v16h a;
#pragma unroll
  for (int v = 0; v < 8; ++v) {
    int k = (v < 4) ? (kb + 2 * v) : (16 + kb + 2 * (v - 4));
    a[2 * v]     = (_Float16)row[k];
    a[2 * v + 1] = (_Float16)row[k + 1];
  }
  return a;
}
// B: 32x16 (KxN) taken from an N-major (transposed) buffer srcT[n][k]:
// lane needs B[k][n] = srcT[n][kof + {kb..kb+7, 16+kb..16+kb+7}], kb = 0/8 -> 2x ds_load_b128.
__device__ __forceinline__ v16h ldsBT(const _Float16* srcT, int ldt, int kof, int nof, int lane) {
  const _Float16* row = srcT + (nof + (lane & 15)) * ldt + kof + ((lane < 16) ? 0 : 8);
  v16h b;
#pragma unroll
  for (int v = 0; v < 8; ++v) {
    int k = (v < 4) ? (2 * v) : (2 * v + 8);
    b[2 * v]     = row[k];
    b[2 * v + 1] = row[k + 1];
  }
  return b;
}

// ================= Fused two-layer GAT, one workgroup (8 waves) per graph =================
__global__ __launch_bounds__(256) void gat_fused_kernel(
    const float* __restrict__ feats, const float* __restrict__ W1,
    const float* __restrict__ a1,    const float* __restrict__ W2,
    const float* __restrict__ a2,    float* __restrict__ extracted)
{
  __shared__ float    xs[NPAD][2];           // zero-padded node features
  __shared__ float    es[NPAD], ed[NPAD];
  __shared__ _Float16 Z1t[NPAD][NPAD];       // per-head Z1 TRANSPOSED: Z1t[o][i]
  __shared__ _Float16 alphaT[NPAD][NPAD];    // alpha^T per head
  __shared__ _Float16 h1s[NPAD][HDIM];       // ELU(h1), f16, row-major [node j][channel d]
  __shared__ float    es2[NPAD], ed2[NPAD], zbar[NPAD];

  const int g    = blockIdx.x;
  const int tid  = threadIdx.x;
  const int lane = tid & 31;
  const int wid  = tid >> 5;

  // prefetch W2 (L2-resident, reused by every block)
  for (int off = tid * 32; off < HDIM * OUTG; off += 256 * 32)
    __builtin_prefetch(&W2[off], 0, 3);

  if (tid < NPAD) { xs[tid][0] = 0.f; xs[tid][1] = 0.f; }
  if (tid < NPAD) { es2[tid] = 0.f; ed2[tid] = 0.f; zbar[tid] = 0.f; }
  __syncthreads();
  for (int idx = tid; idx < N_NODES * 2; idx += 256)
    xs[idx >> 1][idx & 1] = feats[g * (N_NODES * 2) + idx];
  __syncthreads();

  for (int h = 0; h < HEADS; ++h) {
    const float* W1h = W1 + h * (HID * 2);
    const float* a1h = a1 + h * (2 * HID);
    // ---- phase A (parallel halves): threads 0-63 produce Z1t rows (contiguous
    //      b128 stores); threads 64-127 recompute z for es/ed logits ----
    if (tid < NPAD) {
      int   o  = tid;
      float w0 = W1h[o * 2 + 0], w1 = W1h[o * 2 + 1];
#pragma unroll 8
      for (int i = 0; i < NPAD; ++i)
        Z1t[o][i] = (_Float16)(xs[i][0] * w0 + xs[i][1] * w1);
    } else if (tid < 2 * NPAD) {
      int   i  = tid - NPAD;
      float x0 = xs[i][0], x1 = xs[i][1];
      float e_s = 0.f, e_d = 0.f;
#pragma unroll 8
      for (int o = 0; o < HID; ++o) {
        float z = x0 * W1h[o * 2 + 0] + x1 * W1h[o * 2 + 1];
        e_s += z * a1h[o];
        e_d += z * a1h[HID + o];
      }
      es[i] = e_s; ed[i] = e_d;   // rows >= 51 are exactly 0
    }
    __syncthreads();
    // ---- softmax over source i, write alpha^T row j (f16, contiguous) ----
    if (tid < NPAD) {
      int j = tid;
      if (j < N_NODES) {
        float edj = ed[j];
        float mx  = -1e30f;
        for (int i = 0; i < N_NODES; ++i)
          if (i != j) mx = fmaxf(mx, leakyf(es[i] + edj));
        float s = 0.f;
        for (int i = 0; i < NPAD; ++i) {
          float v = 0.f;
          if (i < N_NODES && i != j) { v = __expf(leakyf(es[i] + edj) - mx); s += v; }
          alphaT[j][i] = (_Float16)v;
        }
        float inv = 1.f / s;
        for (int i = 0; i < N_NODES; ++i)
          alphaT[j][i] = (_Float16)((float)alphaT[j][i] * inv);
      } else {
        for (int i = 0; i < NPAD; ++i) alphaT[j][i] = (_Float16)0.f;
      }
    }
    __syncthreads();
    // ---- h1(:, h*64:h*64+64) = alpha^T @ Z1 via WMMA, ELU fused ----
#pragma unroll
    for (int t = 0; t < 2; ++t) {
      int tile = wid * 2 + t;             // 16 tiles of 16x16
      int mt = tile >> 2, nt = tile & 3;
      v8f c = {};
#pragma unroll
      for (int kt = 0; kt < 2; ++kt) {
        v16h a = ldsA(&alphaT[0][0], NPAD, mt * 16, kt * 32, lane);
        v16h b = ldsBT(&Z1t[0][0],   NPAD, kt * 32, nt * 16, lane);
        c = __builtin_amdgcn_wmma_f32_16x16x32_f16(false, a, false, b, (short)0, c, false, false);
      }
      int n  = nt * 16 + (lane & 15);
      int mb = mt * 16 + ((lane < 16) ? 0 : 8);
#pragma unroll
      for (int r = 0; r < 8; ++r) {
        float v = c[r];
        v = (v > 0.f) ? v : (__expf(v) - 1.f);   // ELU
        h1s[mb + r][h * HID + n] = (_Float16)v;
      }
    }
    __syncthreads();
  }

  // ---- Z2^T = W2 @ h1^T (M=OUTG=128, N=NPAD=64, K=HDIM=256) ----
  // D[m=o][n=j] = Z2[j][o]: each lane's 8 C-elements share node n=j, so the
  // es2/ed2/zbar reductions are lane-private; one xor-fold + 3 atomics total.
  {
    const int ntile = wid & 3;            // waves {0,4},{1,5},{2,6},{3,7} pair per ntile
    const int mt0   = (wid >> 2) * 4;     // each wave covers 4 of 8 m-tiles
    const int n     = ntile * 16 + (lane & 15);
    float vs = 0.f, vd = 0.f, vz = 0.f;
#pragma unroll
    for (int mi = 0; mi < 4; ++mi) {
      int mt = mt0 + mi;
      v8f c = {};
      for (int kt = 0; kt < 8; ++kt) {
        v16h a = ldgA(W2, HDIM, mt * 16, kt * 32, lane);              // W2[o][d]
        v16h b = ldsBT(&h1s[0][0], HDIM, kt * 32, ntile * 16, lane);  // h1^T
        c = __builtin_amdgcn_wmma_f32_16x16x32_f16(false, a, false, b, (short)0, c, false, false);
      }
      int mb = mt * 16 + ((lane < 16) ? 0 : 8);
#pragma unroll
      for (int r = 0; r < 8; ++r) {
        float v = c[r];                   // Z2[node n][channel mb+r]
        vs += v * a2[mb + r];
        vd += v * a2[OUTG + mb + r];
        vz += v;
      }
    }
    vz *= (1.f / 128.f);
    vs += __shfl_xor(vs, 16, 32);         // lanes L and L^16 share n
    vd += __shfl_xor(vd, 16, 32);
    vz += __shfl_xor(vz, 16, 32);
    if (lane < 16) {
      atomicAdd(&es2[n],  vs);
      atomicAdd(&ed2[n],  vd);
      atomicAdd(&zbar[n], vz);
    }
  }
  __syncthreads();

  // ---- layer-2 softmax over i, extracted[j] = sum_i alpha2[i,j]*zbar[i] ----
  if (tid < N_NODES) {
    int   j   = tid;
    float edj = ed2[j];
    float mx  = -1e30f;
    for (int i = 0; i < N_NODES; ++i)
      if (i != j) mx = fmaxf(mx, leakyf(es2[i] + edj));
    float s = 0.f, acc = 0.f;
    for (int i = 0; i < N_NODES; ++i)
      if (i != j) {
        float v = __expf(leakyf(es2[i] + edj) - mx);
        s += v; acc += v * zbar[i];
      }
    extracted[g * N_NODES + j] = acc / s;
  }
}

// ================= Two-layer LSTM (seq over B=256, batch F=50) + FC, one block =================
__global__ __launch_bounds__(512) void lstm_fc_kernel(
    const float* __restrict__ extracted,
    const float* __restrict__ Wih0, const float* __restrict__ Whh0,
    const float* __restrict__ bih0, const float* __restrict__ bhh0,
    const float* __restrict__ Wih1, const float* __restrict__ Whh1,
    const float* __restrict__ bih1, const float* __restrict__ bhh1,
    const float* __restrict__ Wfc,  const float* __restrict__ bfc,
    float* __restrict__ out)
{
  __shared__ float xt[LSTM_B][NIN];
  __shared__ float h0[NST], c0[NST], h1[NST], c1[NST];
  const int tid = threadIdx.x;

  for (int p = tid; p < NST; p += 512) { h0[p] = 0.f; c0[p] = 0.f; h1[p] = 0.f; c1[p] = 0.f; }
  __syncthreads();

  for (int t = 0; t < SEQ; ++t) {
    for (int idx = tid; idx < LSTM_B * NIN; idx += 512)
      xt[idx / NIN][idx % NIN] = extracted[t * (LSTM_B * NIN) + idx];
    __syncthreads();

    float nh0[4], nc0[4];
#pragma unroll
    for (int it = 0; it < 4; ++it) {
      int p = tid + it * 512;
      if (p < NST) {
        int f = p >> 5, j = p & 31;
        float pre[4];
#pragma unroll
        for (int gi = 0; gi < 4; ++gi) {
          int row = gi * HSZ + j;
          float s = bih0[row] + bhh0[row];
          const float* wi = Wih0 + row * NIN;
          for (int m = 0; m < NIN; ++m) s += wi[m] * xt[f][m];
          const float* wh = Whh0 + row * HSZ;
          for (int m = 0; m < HSZ; ++m) s += wh[m] * h0[f * HSZ + m];
          pre[gi] = s;
        }
        float cN = sigf(pre[1]) * c0[p] + sigf(pre[0]) * tanhf(pre[2]);
        nc0[it] = cN;
        nh0[it] = sigf(pre[3]) * tanhf(cN);
      }
    }
    __syncthreads();
#pragma unroll
    for (int it = 0; it < 4; ++it) {
      int p = tid + it * 512;
      if (p < NST) { h0[p] = nh0[it]; c0[p] = nc0[it]; }
    }
    __syncthreads();

    float nh1[4], nc1[4];
#pragma unroll
    for (int it = 0; it < 4; ++it) {
      int p = tid + it * 512;
      if (p < NST) {
        int f = p >> 5, j = p & 31;
        float pre[4];
#pragma unroll
        for (int gi = 0; gi < 4; ++gi) {
          int row = gi * HSZ + j;
          float s = bih1[row] + bhh1[row];
          const float* wi = Wih1 + row * HSZ;
          for (int m = 0; m < HSZ; ++m) s += wi[m] * h0[f * HSZ + m];
          const float* wh = Whh1 + row * HSZ;
          for (int m = 0; m < HSZ; ++m) s += wh[m] * h1[f * HSZ + m];
          pre[gi] = s;
        }
        float cN = sigf(pre[1]) * c1[p] + sigf(pre[0]) * tanhf(pre[2]);
        nc1[it] = cN;
        nh1[it] = sigf(pre[3]) * tanhf(cN);
      }
    }
    __syncthreads();
#pragma unroll
    for (int it = 0; it < 4; ++it) {
      int p = tid + it * 512;
      if (p < NST) { h1[p] = nh1[it]; c1[p] = nc1[it]; }
    }
    __syncthreads();

    // FC for this step: wave w -> class w, wave32 shuffle reduce
    if (tid < 256) {
      int cls = tid >> 5, ln = tid & 31;
      float s = 0.f;
      const float* wf = Wfc + cls * NST;
      for (int q = 0; q < 50; ++q) {
        int p = ln * 50 + q;
        s += h1[p] * wf[p];
      }
#pragma unroll
      for (int off = 16; off > 0; off >>= 1) s += __shfl_down(s, off, 32);
      if (ln == 0) out[t * 8 + cls] = s + bfc[cls];
    }
    __syncthreads();
  }
}

extern "C" void kernel_launch(void* const* d_in, const int* in_sizes, int n_in,
                              void* d_out, int out_size, void* d_ws, size_t ws_size,
                              hipStream_t stream) {
  const float* feats = (const float*)d_in[0];
  const float* W1    = (const float*)d_in[1];
  const float* a1    = (const float*)d_in[2];
  const float* W2    = (const float*)d_in[3];
  const float* a2    = (const float*)d_in[4];
  const float* Wih0  = (const float*)d_in[5];
  const float* Whh0  = (const float*)d_in[6];
  const float* bih0  = (const float*)d_in[7];
  const float* bhh0  = (const float*)d_in[8];
  const float* Wih1  = (const float*)d_in[9];
  const float* Whh1  = (const float*)d_in[10];
  const float* bih1  = (const float*)d_in[11];
  const float* bhh1  = (const float*)d_in[12];
  const float* Wfc   = (const float*)d_in[13];
  const float* bfc   = (const float*)d_in[14];
  float* extracted = (float*)d_ws;   // 256*50*51 floats = 2.6 MB
  float* out       = (float*)d_out;  // (256, 8)

  gat_fused_kernel<<<12800, 256, 0, stream>>>(feats, W1, a1, W2, a2, extracted);
  lstm_fc_kernel<<<1, 512, 0, stream>>>(extracted, Wih0, Whh0, bih0, bhh0,
                                        Wih1, Whh1, bih1, bhh1, Wfc, bfc, out);
}
